// SparseMSDeformableAttention_13984413516397
// MI455X (gfx1250) — matrix operand order repaired and checked by reading, not compiled, not run on hardware
//
#include <hip/hip_runtime.h>
#include <hip/hip_bf16.h>

typedef __attribute__((ext_vector_type(16))) _Float16 v16h;
typedef __attribute__((ext_vector_type(8)))  float    v8f;

#define QN    8192
#define CN    256
#define LN    4
#define PN    4
#define NHN   8
#define DHN   32
#define HMAXN 128
#define MVAL  131072      // B*HMAX*HMAX*L rows of the value tensor
#define NQOUT 384         // 256 (offsets) + 128 (attention logits)

// ---------------------------------------------------------------------------
// Weight prep: convert f32 weights to f16 (row-major [N][K]) and build the
// concatenated query-side weight/bias (W_so rows 0..255, W_aw rows 256..383).
// ---------------------------------------------------------------------------
__global__ void prep_weights_kernel(const float* __restrict__ Wv,
                                    const float* __restrict__ Wso,
                                    const float* __restrict__ Waw,
                                    const float* __restrict__ bso,
                                    const float* __restrict__ baw,
                                    const float* __restrict__ Wo,
                                    _Float16* __restrict__ wv16,
                                    _Float16* __restrict__ wq16,
                                    _Float16* __restrict__ wo16,
                                    float* __restrict__ bq) {
  int i = blockIdx.x * blockDim.x + threadIdx.x;
  if (i < 65536) {                       // 256x256 matrices
    wv16[i] = (_Float16)Wv[i];
    wo16[i] = (_Float16)Wo[i];
    wq16[i] = (_Float16)Wso[i];
  }
  if (i < 32768) {                       // 128x256 W_aw appended
    wq16[65536 + i] = (_Float16)Waw[i];
  }
  if (i < 256)                 bq[i] = bso[i];
  else if (i < NQOUT)          bq[i] = baw[i - 256];
}

// ---------------------------------------------------------------------------
// WMMA GEMM with 4x A-fragment reuse:
//   C[M,N] = A[M,K](f32, cvt->f16 on the fly) * Bw[N,K]^T + bias
// One wave computes a 16(M) x 64(N) strip: per K-step it converts the A
// fragment once and issues 4x v_wmma_f32_16x16x32_f16 against 4 B fragments.
// Requires M%16==0, N%64==0, K%32==0 (all hold) -> EXEC all-ones, no guards.
// ---------------------------------------------------------------------------
template <bool OUT_HALF>
__global__ void gemm_wmma_kernel(const float* __restrict__ A,
                                 const _Float16* __restrict__ Bw,
                                 const float* __restrict__ bias,
                                 void* __restrict__ Cout,
                                 int M, int N, int K) {
  const int lane = threadIdx.x & 31;
  const int wave = threadIdx.x >> 5;
  const int n64tiles = N >> 6;                 // 64-wide N tiles
  const int tile = blockIdx.x * 4 + wave;
  const int mt = tile / n64tiles;              // 16-row M tile
  const int nq = tile - mt * n64tiles;         // 64-col N tile

  const int hi = lane >> 4;                    // 0: lanes 0-15, 1: lanes 16-31
  const int m  = (mt << 4) + (lane & 15);      // A row held by this lane
  const int n0 = (nq << 6) + (lane & 15);      // first B column held by lane

  // A layout (16-bit A 16x32): lane holds contiguous 16-element K chunk at base hi*16
  const float* arow = A + (size_t)m * K + hi * 16;
  // B layout: lane holds contiguous 16-half K chunk at base hi*16 (32B loads)
  const _Float16* brow = Bw + (size_t)n0 * K + hi * 16;
  const size_t bstride = (size_t)16 * K;       // next 16-wide N tile

  v8f acc[4] = {};
  for (int k0 = 0; k0 < K; k0 += 32) {
    v16h af;
#pragma unroll
    for (int j = 0; j < 8; ++j) {
      af[j]     = (_Float16)arow[k0 + j];
      af[j + 8] = (_Float16)arow[k0 + 8 + j];
    }
#pragma unroll
    for (int t = 0; t < 4; ++t) {
      v16h bf = *(const v16h*)(brow + (size_t)t * bstride + k0);
      acc[t] = __builtin_amdgcn_wmma_f32_16x16x32_f16(
          /*neg_a=*/false, af, /*neg_b=*/false, bf,
          /*c_mod=*/(short)0, acc[t], /*reuse_a=*/false, /*reuse_b=*/false);
    }
  }

  // C/D layout: VGPR i -> row (mt*16 + i + hi*8), col n
#pragma unroll
  for (int t = 0; t < 4; ++t) {
    const int n = n0 + t * 16;
    const float bv = bias ? bias[n] : 0.0f;
#pragma unroll
    for (int i = 0; i < 8; ++i) {
      const int    row = (mt << 4) + i + hi * 8;
      const size_t idx = (size_t)row * N + n;
      const float  r   = acc[t][i] + bv;
      if (OUT_HALF) ((_Float16*)Cout)[idx] = (_Float16)r;
      else          ((float*)Cout)[idx]    = r;
    }
  }
}

// ---------------------------------------------------------------------------
// Deformable sampling: one wave per (q, head); lane = channel d (DH == 32).
// Reads projected values (f16), offsets/logits (f32), accumulates f32.
// ---------------------------------------------------------------------------
__global__ void msda_sample_kernel(const _Float16* __restrict__ vh,   // [131072][256] f16
                                   const float* __restrict__ outq,    // [Q][384] f32
                                   const float* __restrict__ ref,     // [Q][2] f32
                                   const int* __restrict__ bo,        // batch_offsets[2]
                                   const int* __restrict__ ss,        // spatial_shapes[4][2]
                                   float* __restrict__ attn) {        // [Q][256] f32
  const int task = blockIdx.x * 8 + (threadIdx.x >> 5);
  const int q = task >> 3;
  const int h = task & 7;
  const int d = threadIdx.x & 31;
  const int bq = (q >= bo[1]) ? 1 : 0;

  const float* qrow = outq + (size_t)q * NQOUT;

  // softmax over the 16 (l,p) logits for this (q,h)
  float lg[16];
  float mx = -1e30f;
#pragma unroll
  for (int i = 0; i < 16; ++i) {
    lg[i] = qrow[256 + i * NHN + h];
    mx = fmaxf(mx, lg[i]);
  }
  float s = 0.0f;
#pragma unroll
  for (int i = 0; i < 16; ++i) { lg[i] = __expf(lg[i] - mx); s += lg[i]; }
  const float invs = 1.0f / s;

  // inverse-sigmoid of the reference point (clipped, eps = 1e-5)
  float rx = fminf(fmaxf(ref[2 * q + 0], 0.0f), 1.0f);
  float ry = fminf(fmaxf(ref[2 * q + 1], 0.0f), 1.0f);
  const float ix = __logf(fmaxf(rx, 1e-5f) / fmaxf(1.0f - rx, 1e-5f));
  const float iy = __logf(fmaxf(ry, 1e-5f) / fmaxf(1.0f - ry, 1e-5f));

  const size_t chan = (size_t)h * DHN + d;
  float acc = 0.0f;

#pragma unroll
  for (int l = 0; l < LN; ++l) {
    const int   Hl = ss[l * 2 + 0];
    const int   Wl = ss[l * 2 + 1];
    const float Hf = (float)Hl, Wf = (float)Wl;
    for (int p = 0; p < PN; ++p) {
      const int i = l * PN + p;
      const float ox = qrow[(i * NHN + h) * 2 + 0];
      const float oy = qrow[(i * NHN + h) * 2 + 1];
      // sigmoid(inv_sigmoid(ref) + off); (g+1)*W*0.5-0.5 with g=2*loc-1 == loc*W-0.5
      const float lx = 1.0f / (1.0f + __expf(-(ix + ox)));
      const float ly = 1.0f / (1.0f + __expf(-(iy + oy)));
      const float xf = lx * Wf - 0.5f;
      const float yf = ly * Hf - 0.5f;
      const float x0f = floorf(xf), y0f = floorf(yf);
      const float wx1 = xf - x0f, wx0 = 1.0f - wx1;
      const float wy1 = yf - y0f, wy0 = 1.0f - wy1;
      const int x0 = (int)x0f, y0 = (int)y0f;
      const int x1 = x0 + 1,  y1 = y0 + 1;
      const float aw = lg[i] * invs;

      auto corner = [&](int yi, int xi) -> float {
        const bool valid = (xi >= 0) & (xi < Wl) & (yi >= 0) & (yi < Hl);
        const int xc = min(max(xi, 0), HMAXN - 1);
        const int yc = min(max(yi, 0), HMAXN - 1);
        const size_t row = (((size_t)(bq * HMAXN + yc) * HMAXN) + xc) * LN + l;
        const float val = (float)vh[row * CN + chan];
        return valid ? val : 0.0f;
      };

      const float c00 = corner(y0, x0);
      const float c01 = corner(y0, x1);
      const float c10 = corner(y1, x0);
      const float c11 = corner(y1, x1);
      acc += aw * (wy0 * (wx0 * c00 + wx1 * c01) + wy1 * (wx0 * c10 + wx1 * c11));
    }
  }
  attn[(size_t)q * CN + chan] = acc;
}

// ---------------------------------------------------------------------------
extern "C" void kernel_launch(void* const* d_in, const int* in_sizes, int n_in,
                              void* d_out, int out_size, void* d_ws, size_t ws_size,
                              hipStream_t stream) {
  // setup_inputs() order:
  const float* query  = (const float*)d_in[0];
  const float* refpts = (const float*)d_in[1];
  const float* values = (const float*)d_in[2];
  const float* W_so   = (const float*)d_in[3];
  const float* b_so   = (const float*)d_in[4];
  const float* W_aw   = (const float*)d_in[5];
  const float* b_aw   = (const float*)d_in[6];
  const float* W_v    = (const float*)d_in[7];
  const float* b_v    = (const float*)d_in[8];
  const float* W_o    = (const float*)d_in[9];
  const float* b_o    = (const float*)d_in[10];
  const int*   boffs  = (const int*)d_in[11];
  const int*   sshape = (const int*)d_in[12];
  float*       out    = (float*)d_out;

  // workspace layout (256B-aligned slices)
  char* ws = (char*)d_ws;
  size_t off = 0;
  auto alloc = [&](size_t bytes) -> char* {
    char* p = ws + off;
    off = (off + bytes + 255) & ~(size_t)255;
    return p;
  };
  _Float16* v16  = (_Float16*)alloc((size_t)MVAL * CN * sizeof(_Float16)); // 64 MB projected values
  _Float16* wv16 = (_Float16*)alloc((size_t)CN * CN * sizeof(_Float16));
  _Float16* wq16 = (_Float16*)alloc((size_t)NQOUT * CN * sizeof(_Float16));
  _Float16* wo16 = (_Float16*)alloc((size_t)CN * CN * sizeof(_Float16));
  float*    bq   = (float*)alloc((size_t)NQOUT * sizeof(float));
  float*    outq = (float*)alloc((size_t)QN * NQOUT * sizeof(float));      // 12 MB
  float*    attn = (float*)alloc((size_t)QN * CN * sizeof(float));         // 8 MB

  // 1) convert weights to f16
  prep_weights_kernel<<<(98304 + 255) / 256, 256, 0, stream>>>(
      W_v, W_so, W_aw, b_so, b_aw, W_o, wv16, wq16, wo16, bq);

  // 2) value projection: v16 = values @ W_v^T + b_v   (131072x256x256, f16 out)
  {
    const int waves = (MVAL / 16) * (CN / 64);   // 32768 waves (16x64 strips)
    gemm_wmma_kernel<true><<<waves / 4, 128, 0, stream>>>(
        values, wv16, b_v, (void*)v16, MVAL, CN, CN);
  }

  // 3) fused query projections: outq = query @ [W_so;W_aw]^T + [b_so;b_aw]
  {
    const int waves = (QN / 16) * (NQOUT / 64);  // 3072 waves
    gemm_wmma_kernel<false><<<waves / 4, 128, 0, stream>>>(
        query, wq16, bq, (void*)outq, QN, NQOUT, CN);
  }

  // 4) softmax + deformable bilinear sampling -> attn[Q][C]
  msda_sample_kernel<<<(QN * NHN) / 8, 256, 0, stream>>>(
      v16, outq, refpts, boffs, sshape, attn);

  // 5) output projection: out = attn @ W_o^T + b_o
  {
    const int waves = (QN / 16) * (CN / 64);     // 2048 waves
    gemm_wmma_kernel<false><<<waves / 4, 128, 0, stream>>>(
        attn, wo16, b_o, (void*)out, QN, CN, CN);
  }
}
